// CS_MAMBA_40269613367796
// MI455X (gfx1250) — matrix operand order repaired and checked by reading, not compiled, not run on hardware
//
#include <hip/hip_runtime.h>
#include <hip/hip_bf16.h>
#include <math.h>

// ---------------- types ----------------
typedef __bf16 bf16;
typedef __attribute__((ext_vector_type(16))) __bf16 v16bf;
typedef __attribute__((ext_vector_type(8)))  __bf16 v8bf;
typedef __attribute__((ext_vector_type(4)))  __bf16 v4bf;
typedef __attribute__((ext_vector_type(8)))  float  v8f;
typedef __attribute__((ext_vector_type(4)))  int    v4i;

#define EPSF 1e-5f
// problem constants
#define BATCH 64
#define CCH   2048
#define HH    24
#define WW    12
#define SPAT  288     // H*W
#define LSEQ  12
#define DI    256
#define DSTATE 16

__device__ __forceinline__ v16bf join8(v8bf lo, v8bf hi) {
  return __builtin_shufflevector(lo, hi, 0,1,2,3,4,5,6,7,8,9,10,11,12,13,14,15);
}

// 16-byte global->LDS copy: async DMA path (ASYNCcnt) when available, else VGPR bounce.
__device__ __forceinline__ void copy16_g2l(const bf16* __restrict__ g, bf16* __restrict__ l) {
#if defined(__gfx1250__) && __has_builtin(__builtin_amdgcn_global_load_async_to_lds_b128)
  typedef __attribute__((address_space(1))) v4i* gp_t;
  typedef __attribute__((address_space(3))) v4i* lp_t;
  __builtin_amdgcn_global_load_async_to_lds_b128((gp_t)g, (lp_t)l, 0, 0);
#else
  *(v8bf*)l = *(const v8bf*)g;
#endif
}

__device__ __forceinline__ void async_wait_all() {
#if defined(__gfx1250__) && __has_builtin(__builtin_amdgcn_s_wait_asynccnt)
  __builtin_amdgcn_s_wait_asynccnt(0);
#elif defined(__gfx1250__) && __has_builtin(__builtin_amdgcn_global_load_async_to_lds_b128)
  asm volatile("s_wait_asynccnt 0x0" ::: "memory");
#endif
}

// ---------------- 0) fp32 -> bf16 weight pre-conversion ----------------
__global__ void cvt_f32_bf16_kernel(const float* __restrict__ src, bf16* __restrict__ dst, int n4) {
  int t = blockIdx.x * 256 + threadIdx.x;
  if (t < n4) {
    float4 f = *(const float4*)(src + (size_t)t * 4);
    v4bf p = { (bf16)f.x, (bf16)f.y, (bf16)f.z, (bf16)f.w };
    *(v4bf*)(dst + (size_t)t * 4) = p;
  }
}

// ---------------- 1) pooling + interleave into vi (B, 12, C) ----------------
__global__ void pool_vi_kernel(const float* __restrict__ vis, const float* __restrict__ inf,
                               float* __restrict__ vi) {
  int t = blockIdx.x * blockDim.x + threadIdx.x;       // B*C*12 threads
  if (t >= BATCH * CCH * LSEQ) return;
  int l = t % LSEQ;
  int c = (t / LSEQ) & (CCH - 1);
  int b = t / (LSEQ * CCH);
  int p = l >> 1, mod = l & 1;
  const float* src  = mod ? inf : vis;
  const float* base = src + ((size_t)(b * CCH + c) * HH + p * 4) * WW; // 48 contiguous floats
  float s = 0.f;
#pragma unroll
  for (int i = 0; i < 12; i++) {
    float4 f = *(const float4*)(base + i * 4);
    s += f.x + f.y + f.z + f.w;
  }
  vi[((size_t)b * LSEQ + l) * CCH + c] = s * (1.f / 48.f);
}

// ---------------- 2) LayerNorm over C=2048, one block per row ----------------
__global__ void layernorm_kernel(const float* __restrict__ in, const float* __restrict__ g,
                                 const float* __restrict__ bb, float* __restrict__ out) {
  int row = blockIdx.x;
  const float* x = in + (size_t)row * CCH;
  float* o = out + (size_t)row * CCH;
  __shared__ float red[256];
  float s = 0.f;
  for (int c = threadIdx.x; c < CCH; c += 256) s += x[c];
  red[threadIdx.x] = s; __syncthreads();
  for (int st = 128; st > 0; st >>= 1) { if (threadIdx.x < st) red[threadIdx.x] += red[threadIdx.x + st]; __syncthreads(); }
  float mean = red[0] * (1.f / CCH);
  __syncthreads();
  float v = 0.f;
  for (int c = threadIdx.x; c < CCH; c += 256) { float d = x[c] - mean; v += d * d; }
  red[threadIdx.x] = v; __syncthreads();
  for (int st = 128; st > 0; st >>= 1) { if (threadIdx.x < st) red[threadIdx.x] += red[threadIdx.x + st]; __syncthreads(); }
  float rstd = rsqrtf(red[0] * (1.f / CCH) + EPSF);
  for (int c = threadIdx.x; c < CCH; c += 256) o[c] = (x[c] - mean) * rstd * g[c] + bb[c];
}

// ---------------- 3) generic WMMA GEMM: D(MxN,ldd) = A(MxK,lda) * Bw(NxK)^T (+bias_n)(+addsrc) ----
// 128 threads = 4 waves; block tile 64x64; wave tile 32x32 = 2x2 fragments of 16x16.
// Branchless float4 staging: all loads unconditional (clamped addresses), zeros by select.
__global__ void wmma_gemm_nt(const float* __restrict__ A, int lda,
                             const float* __restrict__ Bw,
                             const float* __restrict__ bias,
                             const float* __restrict__ addsrc,
                             float* __restrict__ D, int ldd,
                             int M, int N, int K) {
  __shared__ __align__(16) bf16 sA[64][32];
  __shared__ __align__(16) bf16 sB[64][32];
  int m0 = blockIdx.x * 64, n0 = blockIdx.y * 64;
  int tid = threadIdx.x;
  int wid = tid >> 5, lane = tid & 31;
  int wm = (wid >> 1) * 32, wn = (wid & 1) * 32;
  int half = lane >> 4, l16 = lane & 15;

  v8f acc[2][2];
  v8f z = {0.f,0.f,0.f,0.f,0.f,0.f,0.f,0.f};
#pragma unroll
  for (int i = 0; i < 2; i++)
#pragma unroll
    for (int j = 0; j < 2; j++) acc[i][j] = z;

  for (int k0 = 0; k0 < K; k0 += 32) {
#pragma unroll
    for (int i = tid; i < 512; i += 128) {
      int r  = i >> 3;
      int k4 = (i & 7) * 4;
      int ka = k0 + k4;
      int kc = (ka < K) ? ka : 0;

      int rm = m0 + r;
      int rmc = (rm < M) ? rm : (M - 1);
      float selA = ((rm < M) && (ka < K)) ? 1.f : 0.f;
      float4 av = *(const float4*)(A + (size_t)rmc * lda + kc);
      v4bf ap = { (bf16)(av.x * selA), (bf16)(av.y * selA),
                  (bf16)(av.z * selA), (bf16)(av.w * selA) };
      *(v4bf*)&sA[r][k4] = ap;

      int rn = n0 + r;
      int rnc = (rn < N) ? rn : (N - 1);
      float selB = ((rn < N) && (ka < K)) ? 1.f : 0.f;
      float4 bv = *(const float4*)(Bw + (size_t)rnc * K + kc);
      v4bf bp = { (bf16)(bv.x * selB), (bf16)(bv.y * selB),
                  (bf16)(bv.z * selB), (bf16)(bv.w * selB) };
      *(v4bf*)&sB[r][k4] = bp;
    }
    __syncthreads();
    v16bf aF[2], bF[2];
#pragma unroll
    for (int i = 0; i < 2; i++) {
      const bf16* pa = &sA[wm + i * 16 + l16][0];
      aF[i] = join8(*(const v8bf*)(pa + half * 8), *(const v8bf*)(pa + half * 8 + 16));
      const bf16* pb = &sB[wn + i * 16 + l16][0];
      bF[i] = join8(*(const v8bf*)(pb + half * 8), *(const v8bf*)(pb + half * 8 + 16));
    }
#pragma unroll
    for (int i = 0; i < 2; i++)
#pragma unroll
      for (int j = 0; j < 2; j++)
        acc[i][j] = __builtin_amdgcn_wmma_f32_16x16x32_bf16(
            false, aF[i], false, bF[j], (short)0, acc[i][j], false, false);
    __syncthreads();
  }

#pragma unroll
  for (int i = 0; i < 2; i++)
#pragma unroll
    for (int j = 0; j < 2; j++) {
      int mbase = m0 + wm + i * 16 + half * 8;
      int n = n0 + wn + j * 16 + l16;
      if (n < N) {
        float badd = bias ? bias[n] : 0.f;
#pragma unroll
        for (int r = 0; r < 8; r++) {
          int m = mbase + r;
          if (m < M) {
            float v = acc[i][j][r] + badd;
            if (addsrc) v += addsrc[(size_t)m * ldd + n];
            D[(size_t)m * ldd + n] = v;
          }
        }
      }
    }
}

// ---------------- 4) depthwise conv3 (over L) + SiLU ----------------
__global__ void conv_silu_kernel(const float* __restrict__ x, const float* __restrict__ w,
                                 const float* __restrict__ cb, float* __restrict__ o) {
  int t = blockIdx.x * 256 + threadIdx.x;       // B*12*256
  if (t >= BATCH * LSEQ * DI) return;
  int d = t & (DI - 1);
  int l = (t >> 8) % LSEQ;
  int b = t / (LSEQ * DI);
  const float* xb = x + (size_t)b * LSEQ * DI;
  float acc = cb[d];
#pragma unroll
  for (int j = 0; j < 3; j++) {
    int ll = l + j - 1;
    if (ll >= 0 && ll < LSEQ) acc += w[d * 3 + j] * xb[ll * DI + d];
  }
  o[t] = acc / (1.f + __expf(-acc));            // silu
}

// ---------------- 5) softplus elementwise ----------------
__global__ void softplus_kernel(float* __restrict__ x, int n) {
  int t = blockIdx.x * 256 + threadIdx.x;
  if (t < n) {
    float v = x[t];
    x[t] = (v > 20.f) ? v : log1pf(__expf(v));
  }
}

// ---------------- 6) selective scan: block per batch, thread per channel ----------------
__global__ void scan_kernel(const float* __restrict__ delta, const float* __restrict__ xs,
                            const float* __restrict__ dbc, const float* __restrict__ A_log,
                            const float* __restrict__ D_ssm, float* __restrict__ y) {
  int b = blockIdx.x, d = threadIdx.x;          // 256 threads
  float Arow[DSTATE];
#pragma unroll
  for (int n = 0; n < DSTATE; n++) Arow[n] = -__expf(A_log[d * DSTATE + n]);
  float Dd = D_ssm[d];
  float h[DSTATE];
#pragma unroll
  for (int n = 0; n < DSTATE; n++) h[n] = 0.f;
  for (int l = 0; l < LSEQ; l++) {
    size_t bl = (size_t)(b * LSEQ + l);
    float dt = delta[bl * DI + d];
    float xv = xs[bl * DI + d];
    const float* bc = dbc + bl * 48;
    float yv = xv * Dd;
#pragma unroll
    for (int n = 0; n < DSTATE; n++) {
      float Bn = bc[16 + n], Cn = bc[32 + n];
      h[n] = __expf(dt * Arow[n]) * h[n] + dt * Bn * xv;
      yv += h[n] * Cn;
    }
    y[bl * DI + d] = yv;
  }
}

// ---------------- 7) channel attention: block per (b, modality), 128 threads ----------------
__global__ void chan_att_kernel(const float* __restrict__ vi2,
                                const float* __restrict__ w1, const float* __restrict__ bn_g,
                                const float* __restrict__ bn_b, const float* __restrict__ w2,
                                float* __restrict__ a_out) {  // (2, B, C)
  int b = blockIdx.x, mod = blockIdx.y;
  __shared__ float sm[CCH], sx[CCH];
  __shared__ float hsum[128];
  const float* base = vi2 + (size_t)b * LSEQ * CCH;
  for (int c = threadIdx.x; c < CCH; c += 128) {
    float mn = 0.f, mx = -1e30f;
    for (int p = 0; p < 6; p++) {
      float v = base[(2 * p + mod) * CCH + c];
      mn += v; mx = fmaxf(mx, v);
    }
    sm[c] = mn * (1.f / 6.f); sx[c] = mx;
  }
  __syncthreads();
  {
    int hh = threadIdx.x;
    const float* wr = w1 + (size_t)hh * CCH;
    float am = 0.f, ax = 0.f;
    for (int c = 0; c < CCH; c++) { float w = wr[c]; am += w * sm[c]; ax += w * sx[c]; }
    float gs = bn_g[hh] * rsqrtf(1.f + EPSF);
    am = fmaxf(0.f, am * gs + bn_b[hh]);
    ax = fmaxf(0.f, ax * gs + bn_b[hh]);
    hsum[hh] = am + ax;                 // shared w2 => fc(mean)+fc(max) collapses
  }
  __syncthreads();
  float* aout = a_out + ((size_t)mod * BATCH + b) * CCH;
  for (int c = threadIdx.x; c < CCH; c += 128) {
    const float* w2r = w2 + (size_t)c * 128;
    float s = 0.f;
    for (int h = 0; h < 128; h++) s += w2r[h] * hsum[h];
    aout[c] = 1.f / (1.f + __expf(-s));
  }
}

// ---------------- 8) big FFN GEMM (the roofline-dominant kernel) ----------------
// out[mod][b][o][s] = relu(bn(Sum_c W[o,c]*(a[b,c]*fm[b,c,s]) + bias[o]))
// Weights pre-converted to bf16 in workspace: A-tile staging is a pure copy ->
// async global->LDS DMA (ASYNCcnt) when available.
// 256 threads = 8 waves (4x2); block tile 128(o) x 96(s); K-tile 64.
__global__ void ffn_wmma_kernel(const float* __restrict__ vis, const float* __restrict__ inf,
                                const float* __restrict__ a_all,
                                const bf16* __restrict__ wvb, const bf16* __restrict__ wib,
                                const float* __restrict__ bv,
                                const float* __restrict__ gv, const float* __restrict__ bbv,
                                const float* __restrict__ bi,
                                const float* __restrict__ gi, const float* __restrict__ bbi,
                                float* __restrict__ out) {
  int mt = blockIdx.x / 3, nt = blockIdx.x % 3;
  int b = blockIdx.y, mod = blockIdx.z;
  const float* fm = (mod ? inf : vis) + (size_t)b * CCH * SPAT;
  const float* a  = a_all + ((size_t)mod * BATCH + b) * CCH;
  const bf16*  Wcb = mod ? wib : wvb;
  const float* bc = mod ? bi : bv;
  const float* gc = mod ? gi : gv;
  const float* bb = mod ? bbi : bbv;
  float* op = out + (((size_t)mod * BATCH + b) * CCH) * SPAT;

  __shared__ __align__(16) bf16 sA[128][64];   // W tile (o x c), bf16      16 KB
  __shared__ __align__(16) bf16 sB[96][64];    // scaled fm tile (s x c)    12 KB
  __shared__ float sBias[128], sG[128], sBB[128];

  int tid = threadIdx.x;
  int wid = tid >> 5, lane = tid & 31;
  int wm = (wid >> 1) * 32;                    // 0,32,64,96
  int wn = (wid & 1) * 48;                     // 0,48
  int half = lane >> 4, l16 = lane & 15;
  int m0 = mt * 128, s0 = nt * 96;

  if (tid < 128) {
    int o = m0 + tid;
    sBias[tid] = bc[o];
    sG[tid]    = gc[o] * rsqrtf(1.f + EPSF);
    sBB[tid]   = bb[o];
  }

  v8f acc[2][3];
  v8f z = {0.f,0.f,0.f,0.f,0.f,0.f,0.f,0.f};
#pragma unroll
  for (int i = 0; i < 2; i++)
#pragma unroll
    for (int j = 0; j < 3; j++) acc[i][j] = z;

  for (int k0 = 0; k0 < CCH; k0 += 64) {
    // stage bf16 weight tile 128x64: 1024 x 16-byte pure copies (async DMA path)
#pragma unroll
    for (int i = tid; i < 1024; i += 256) {
      int r  = i >> 3;
      int k8 = (i & 7) * 8;
      copy16_g2l(Wcb + (size_t)(m0 + r) * CCH + k0 + k8, &sA[r][k8]);
    }
    // stage attention-scaled activation tile transposed to (s, c):
    // 1536 float4 granules, 6 per thread (contiguous along spatial dim)
#pragma unroll
    for (int i = tid; i < 1536; i += 256) {
      int k  = i / 24;
      int s4 = (i % 24) * 4;
      int c = k0 + k;
      float sc = a[c];
      float4 f = *(const float4*)(fm + (size_t)c * SPAT + s0 + s4);
      sB[s4 + 0][k] = (bf16)(sc * f.x);
      sB[s4 + 1][k] = (bf16)(sc * f.y);
      sB[s4 + 2][k] = (bf16)(sc * f.z);
      sB[s4 + 3][k] = (bf16)(sc * f.w);
    }
    // prefetch next weight tile rows (global_prefetch_b8 path)
    if (k0 + 64 < CCH && tid < 128)
      __builtin_prefetch(Wcb + (size_t)(m0 + tid) * CCH + k0 + 64, 0, 1);
    async_wait_all();
    __syncthreads();

#pragma unroll
    for (int kk = 0; kk < 64; kk += 32) {
      v16bf aF[2], bF[3];
#pragma unroll
      for (int i = 0; i < 2; i++) {
        const bf16* pa = &sA[wm + i * 16 + l16][kk];
        aF[i] = join8(*(const v8bf*)(pa + half * 8), *(const v8bf*)(pa + half * 8 + 16));
      }
#pragma unroll
      for (int j = 0; j < 3; j++) {
        const bf16* pb = &sB[wn + j * 16 + l16][kk];
        bF[j] = join8(*(const v8bf*)(pb + half * 8), *(const v8bf*)(pb + half * 8 + 16));
      }
#pragma unroll
      for (int i = 0; i < 2; i++)
#pragma unroll
        for (int j = 0; j < 3; j++)
          acc[i][j] = __builtin_amdgcn_wmma_f32_16x16x32_bf16(
              false, aF[i], false, bF[j], (short)0, acc[i][j], false, false);
    }
    __syncthreads();
  }

  // fused bias + BN + ReLU epilogue
#pragma unroll
  for (int i = 0; i < 2; i++)
#pragma unroll
    for (int j = 0; j < 3; j++) {
      int mloc = wm + i * 16 + half * 8;
      int n = s0 + wn + j * 16 + l16;
#pragma unroll
      for (int r = 0; r < 8; r++) {
        int ml = mloc + r;
        float v = acc[i][j][r] + sBias[ml];
        v = fmaxf(0.f, v * sG[ml] + sBB[ml]);
        op[(size_t)(m0 + ml) * SPAT + n] = v;
      }
    }
}

// ---------------- host launcher ----------------
extern "C" void kernel_launch(void* const* d_in, const int* in_sizes, int n_in,
                              void* d_out, int out_size, void* d_ws, size_t ws_size,
                              hipStream_t stream) {
  const float* vis      = (const float*)d_in[0];
  const float* infm     = (const float*)d_in[1];
  const float* ln1_g    = (const float*)d_in[2];
  const float* ln1_b    = (const float*)d_in[3];
  const float* ln2_g    = (const float*)d_in[4];
  const float* ln2_b    = (const float*)d_in[5];
  const float* W_in     = (const float*)d_in[6];
  const float* conv_w   = (const float*)d_in[7];
  const float* conv_b   = (const float*)d_in[8];
  const float* A_log    = (const float*)d_in[9];
  const float* Wx       = (const float*)d_in[10];
  const float* Wdt      = (const float*)d_in[11];
  const float* bdt      = (const float*)d_in[12];
  const float* D_ssm    = (const float*)d_in[13];
  const float* W_out    = (const float*)d_in[14];
  const float* att_w1   = (const float*)d_in[15];
  const float* att_bn_g = (const float*)d_in[16];
  const float* att_bn_b = (const float*)d_in[17];
  const float* att_w2   = (const float*)d_in[18];
  const float* fv_w     = (const float*)d_in[19];
  const float* fv_b     = (const float*)d_in[20];
  const float* fv_g     = (const float*)d_in[21];
  const float* fv_bb    = (const float*)d_in[22];
  const float* fi_w     = (const float*)d_in[23];
  const float* fi_b     = (const float*)d_in[24];
  const float* fi_g     = (const float*)d_in[25];
  const float* fi_bb    = (const float*)d_in[26];

  float* ws = (float*)d_ws;
  size_t off = 0;
  float* vi    = ws + off; off += (size_t)BATCH * LSEQ * CCH;   // residual / later vi2
  float* xn    = ws + off; off += (size_t)BATCH * LSEQ * CCH;   // LN1 out / pre-LN2
  float* x1    = ws + off; off += (size_t)BATCH * LSEQ * DI;
  float* xs    = ws + off; off += (size_t)BATCH * LSEQ * DI;
  float* dbc   = ws + off; off += (size_t)BATCH * LSEQ * 48;
  float* delta = ws + off; off += (size_t)BATCH * LSEQ * DI;
  float* y     = ws + off; off += (size_t)BATCH * LSEQ * DI;
  float* a     = ws + off; off += (size_t)2 * BATCH * CCH;
  bf16* wv_bf  = (bf16*)(ws + off);                       // 2048x2048 bf16
  bf16* wi_bf  = wv_bf + (size_t)CCH * CCH;               // 2048x2048 bf16
  off += (size_t)CCH * CCH;                               // 2 bf16 mats == CCH*CCH floats
  (void)off; (void)ws_size; (void)in_sizes; (void)n_in; (void)out_size;

  const int R = BATCH * LSEQ;   // 768 rows

  pool_vi_kernel<<<(BATCH * CCH * LSEQ + 255) / 256, 256, 0, stream>>>(vis, infm, vi);
  layernorm_kernel<<<R, 256, 0, stream>>>(vi, ln1_g, ln1_b, xn);

  { dim3 g(12, 4);  wmma_gemm_nt<<<g, 128, 0, stream>>>(xn, CCH, W_in, nullptr, nullptr, x1, DI, R, DI, CCH); }
  conv_silu_kernel<<<(R * DI + 255) / 256, 256, 0, stream>>>(x1, conv_w, conv_b, xs);
  { dim3 g(12, 1);  wmma_gemm_nt<<<g, 128, 0, stream>>>(xs, DI, Wx, nullptr, nullptr, dbc, 48, R, 48, DI); }
  { dim3 g(12, 4);  wmma_gemm_nt<<<g, 128, 0, stream>>>(dbc, 48, Wdt, bdt, nullptr, delta, DI, R, DI, 16); }
  softplus_kernel<<<(R * DI + 255) / 256, 256, 0, stream>>>(delta, R * DI);
  scan_kernel<<<BATCH, DI, 0, stream>>>(delta, xs, dbc, A_log, D_ssm, y);
  { dim3 g(12, 32); wmma_gemm_nt<<<g, 128, 0, stream>>>(y, DI, W_out, nullptr, vi, xn, CCH, R, CCH, DI); }
  layernorm_kernel<<<R, 256, 0, stream>>>(xn, ln2_g, ln2_b, vi);   // vi := vi2

  { dim3 g(BATCH, 2); chan_att_kernel<<<g, 128, 0, stream>>>(vi, att_w1, att_bn_g, att_bn_b, att_w2, a); }

  // pre-convert FFN weights to bf16 (re-read 192x by FFN blocks; halves that traffic)
  cvt_f32_bf16_kernel<<<(CCH * CCH / 4 + 255) / 256, 256, 0, stream>>>(fv_w, wv_bf, CCH * CCH / 4);
  cvt_f32_bf16_kernel<<<(CCH * CCH / 4 + 255) / 256, 256, 0, stream>>>(fi_w, wi_bf, CCH * CCH / 4);

  { dim3 g(48, BATCH, 2);   // 16 m-tiles * 3 n-tiles
    ffn_wmma_kernel<<<g, 256, 0, stream>>>(vis, infm, a,
                                           wv_bf, wi_bf,
                                           fv_b, fv_g, fv_bb,
                                           fi_b, fi_g, fi_bb,
                                           (float*)d_out); }
}
